// SpikeUniAffineAttention_46411416601154
// MI455X (gfx1250) — matrix operand order repaired and checked by reading, not compiled, not run on hardware
//
#include <hip/hip_runtime.h>
#include <hip/hip_bf16.h>

// ---------------------------------------------------------------------------
// SpikeUniAffineAttention for MI455X (gfx1250), wave32 + WMMA f16 (K=32).
// B=2, S=1024, HIDDEN=2048, NH=16, NKV=4, HD=128, LEVEL=2.
// Forward-value algebra: STE spikes == hard quantized values, so the
// "multi" difference products collapse to single GEMMs; attn is binary
// (score>0 under causal mask); q/k/v/o/out are ternary.
// LDS staging uses GLOBAL_LOAD_ASYNC_TO_LDS_B128 (ASYNCcnt) when available.
// ---------------------------------------------------------------------------

typedef __attribute__((ext_vector_type(16))) _Float16 v16h;
typedef __attribute__((ext_vector_type(8)))  _Float16 v8h;
typedef __attribute__((ext_vector_type(8)))  float    v8f;
typedef __attribute__((ext_vector_type(4)))  int      v4i;

union FragH { v16h v; v8h h8[2]; _Float16 h[16]; };

#define LDT 40   // padded LDS tile stride (f16 elems); 80B rows keep 16B align

#if defined(__AMDGCN__) && __has_builtin(__builtin_amdgcn_global_load_async_to_lds_b128)
#define HAVE_ASYNC_LDS 1
#else
#define HAVE_ASYNC_LDS 0
#endif

// 16-byte global -> LDS tile copy. Async DMA path on CDNA5, VGPR fallback.
// Builtin signature (from toolchain diagnostics): (as1 int4*, as3 int4*, Ii, Ii)
__device__ __forceinline__ void cp16_to_lds(_Float16* lds, const _Float16* g) {
#if HAVE_ASYNC_LDS
  __builtin_amdgcn_global_load_async_to_lds_b128(
      (__attribute__((address_space(1))) v4i*)(g),
      (__attribute__((address_space(3))) v4i*)(lds), 0, 0);
#else
  *(v8h*)lds = *(const v8h*)g;
#endif
}

// Wait for this wave's outstanding async LDS-DMA ops (ASYNCcnt == 0).
__device__ __forceinline__ void async_wait0() {
#if HAVE_ASYNC_LDS
#if __has_builtin(__builtin_amdgcn_s_wait_asynccnt)
  __builtin_amdgcn_s_wait_asynccnt(0);
#else
  asm volatile("s_wait_asynccnt 0x0" ::: "memory");
#endif
#endif
}

// A-matrix 16x32 f16 fragment (ISA 7.12.2): lane holds row M=lane%16,
// halves: K = half*8..+7 then K = 16+half*8..+7  (two contiguous 16B chunks)
__device__ __forceinline__ v16h load_a_frag(const _Float16* base, int ldk) {
  const int lane = threadIdx.x & 31;
  const _Float16* p = base + (lane & 15) * ldk + (lane >> 4) * 8;
  FragH f;
  f.h8[0] = *(const v8h*)p;
  f.h8[1] = *(const v8h*)(p + 16);
  return f.v;
}

// B-matrix 32x16 f16 fragment read from a TRANSPOSED tile BT[n][k]:
// lane holds column n=lane%16, K = (lane>>4)*8..+7 then 16+(lane>>4)*8..+7.
__device__ __forceinline__ v16h load_bt_frag(const _Float16* base, int ldk) {
  const int lane = threadIdx.x & 31;
  const _Float16* p = base + (lane & 15) * ldk + (lane >> 4) * 8;
  FragH f;
  f.h8[0] = *(const v8h*)p;
  f.h8[1] = *(const v8h*)(p + 16);
  return f.v;
}

__device__ __forceinline__ v8f wmma_f16(v16h a, v16h b, v8f c) {
  // v_wmma_f32_16x16x32_f16
  return __builtin_amdgcn_wmma_f32_16x16x32_f16(false, a, false, b,
                                                (short)0, c, false, false);
}

// ternary quantizer: clip(rint(x), -1, 1)   (jnp.round == half-to-even == rintf)
__device__ __forceinline__ float tern(float x) {
  return fminf(1.f, fmaxf(-1.f, rintf(x)));
}

// ---------------------------------------------------------------------------
// Kernel 0: f32 -> f16 conversion (x, Wqkv, Wo); n multiple of 2048
// ---------------------------------------------------------------------------
__global__ __launch_bounds__(256)
void cvt_kernel(const float* __restrict__ src, _Float16* __restrict__ dst, int n) {
  const int i = (blockIdx.x * 256 + threadIdx.x) * 8;
  if (i < n) {
    float4 a = *(const float4*)(src + i);
    float4 b = *(const float4*)(src + i + 4);
    v8h o;
    o[0] = (_Float16)a.x; o[1] = (_Float16)a.y;
    o[2] = (_Float16)a.z; o[3] = (_Float16)a.w;
    o[4] = (_Float16)b.x; o[5] = (_Float16)b.y;
    o[6] = (_Float16)b.z; o[7] = (_Float16)b.w;
    *(v8h*)(dst + i) = o;
  }
}

// ---------------------------------------------------------------------------
// Kernel 1: qkv = x @ Wqkv^T  (M=2048, N=3072, K=2048), fused epilogue:
// ternary quantize, RoPE (q/k sections), write QR/KR row-major and VT
// transposed.  grid=(16,24) block=256 (8 waves); WG tile 128x128.
// ---------------------------------------------------------------------------
__global__ __launch_bounds__(256)
void gemm_qkv_rope_kernel(const _Float16* __restrict__ X,
                          const _Float16* __restrict__ W,
                          _Float16* __restrict__ QR,
                          _Float16* __restrict__ KR,
                          _Float16* __restrict__ VT) {
  __shared__ alignas(16) _Float16 As[128 * LDT];
  __shared__ alignas(16) _Float16 Bs[128 * LDT];
  const int tid  = threadIdx.x;
  const int lane = tid & 31, wave = tid >> 5;
  const int mBase = blockIdx.x * 128;
  const int sec   = blockIdx.y;          // 0..15 Q heads, 16..19 K, 20..23 V
  const int nBase = sec * 128;
  const int lrow = tid >> 1;             // cooperative loader: 0..127
  const int lcol = (tid & 1) * 16;       // 0 or 16

  v8f acc[8] = {};

  for (int k0 = 0; k0 < 2048; k0 += 32) {
    __syncthreads();
    {
      const _Float16* gA = X + (size_t)(mBase + lrow) * 2048 + k0 + lcol;
      cp16_to_lds(&As[lrow * LDT + lcol],     gA);
      cp16_to_lds(&As[lrow * LDT + lcol + 8], gA + 8);
      const _Float16* gB = W + (size_t)(nBase + lrow) * 2048 + k0 + lcol;
      cp16_to_lds(&Bs[lrow * LDT + lcol],     gB);
      cp16_to_lds(&Bs[lrow * LDT + lcol + 8], gB + 8);
      if (k0 + 32 < 2048) {              // CDNA5 global_prefetch
        __builtin_prefetch((const void*)(gA + 32), 0, 1);
        __builtin_prefetch((const void*)(gB + 32), 0, 1);
      }
      async_wait0();
    }
    __syncthreads();
    v16h a = load_a_frag(&As[wave * 16 * LDT], LDT);
#pragma unroll
    for (int f = 0; f < 8; ++f) {
      v16h b = load_bt_frag(&Bs[f * 16 * LDT], LDT);
      acc[f] = wmma_f16(a, b, acc[f]);
    }
  }

  // ---- epilogue: ternary quantize, then RoPE / transpose-store ----
  const int n15 = lane & 15, half = lane >> 4;
  float q[8][8];
#pragma unroll
  for (int f = 0; f < 8; ++f)
#pragma unroll
    for (int r = 0; r < 8; ++r)
      q[f][r] = tern(acc[f][r]);

  if (sec < 20) {
    // RoPE: d and d^64 partner live in frag f^4 of the SAME lane.
    float inv4[4];
#pragma unroll
    for (int j = 0; j < 4; ++j)   // 10000^(-(d%64)/64) = exp(-ln(1e4)/64 * d63)
      inv4[j] = __expf(-0.14391156f * (float)(j * 16 + n15));
#pragma unroll
    for (int f = 0; f < 8; ++f) {
      const int d   = f * 16 + n15;
      const float iv = inv4[f & 3];
#pragma unroll
      for (int r = 0; r < 8; ++r) {
        const int m   = wave * 16 + r + 8 * half;
        const int tok = mBase + m;
        const int b   = tok >> 10, s = tok & 1023;
        float sn, c;
        __sincosf((float)s * iv, &sn, &c);
        const float rot = (f < 4) ? -q[f ^ 4][r] : q[f ^ 4][r];
        const float val = q[f][r] * c + rot * sn;
        if (sec < 16)
          QR[(((size_t)(b * 16 + sec)) * 1024 + s) * 128 + d] = (_Float16)val;
        else
          KR[(((size_t)(b * 4 + (sec - 16))) * 1024 + s) * 128 + d] = (_Float16)val;
      }
    }
  } else {
    const int hk = sec - 20;
#pragma unroll
    for (int f = 0; f < 8; ++f) {
      const int d = f * 16 + n15;
#pragma unroll
      for (int r = 0; r < 8; ++r) {
        const int m   = wave * 16 + r + 8 * half;
        const int tok = mBase + m;
        const int b   = tok >> 10, s = tok & 1023;
        VT[(((size_t)(b * 4 + hk)) * 128 + d) * 1024 + s] = (_Float16)q[f][r];
      }
    }
  }
}

// ---------------------------------------------------------------------------
// Kernel 2: causal binary attention + A@V per (b,h).
// grid=(8, 32): x = 128-row query block, y = b*16+h.  block=256.
// ---------------------------------------------------------------------------
__global__ __launch_bounds__(256)
void attention_kernel(const _Float16* __restrict__ QR,
                      const _Float16* __restrict__ KR,
                      const _Float16* __restrict__ VT,
                      _Float16* __restrict__ OQ) {
  __shared__ alignas(16) _Float16 Ks[32 * 136];    // K_rot chunk, BT layout [t][d]
  __shared__ alignas(16) _Float16 Vs[128 * LDT];   // V^T chunk,  BT layout [d][t]
  __shared__ alignas(16) _Float16 Attn[128 * LDT]; // binarized attn [s][t]
  const int tid  = threadIdx.x;
  const int lane = tid & 31, wave = tid >> 5;
  const int n15  = lane & 15, half = lane >> 4;
  const int qBase = blockIdx.x * 128;
  const int bh = blockIdx.y;
  const int b = bh >> 4, h = bh & 15, hk = h >> 2;   // GQA: kv head = h / GROUPS
  const int qm = qBase + wave * 16;                  // this wave's 16 query rows

  // resident Q A-fragments (K = 0..127 in 4 chunks of 32)
  const _Float16* Qbase = QR + ((size_t)bh * 1024 + qm) * 128;
  v16h qf[4];
#pragma unroll
  for (int kc = 0; kc < 4; ++kc)
    qf[kc] = load_a_frag(Qbase + kc * 32, 128);

  v8f oacc[8] = {};
  const _Float16* Kglob = KR + ((size_t)(b * 4 + hk)) * 1024 * 128;
  const _Float16* Vglob = VT + ((size_t)(b * 4 + hk)) * 128 * 1024;
  const int krow = tid >> 3, kcol = (tid & 7) * 16;  // Ks loader: 32 x 128
  const int vrow = tid >> 1, vcol = (tid & 1) * 16;  // Vs loader: 128 x 32
  const int nCh = (qBase >> 5) + 4;                  // causal chunk count

  for (int tc = 0; tc < nCh; ++tc) {
    const int tBase = tc * 32;
    __syncthreads();
    {
      const _Float16* gK = Kglob + (size_t)(tBase + krow) * 128 + kcol;
      cp16_to_lds(&Ks[krow * 136 + kcol],     gK);
      cp16_to_lds(&Ks[krow * 136 + kcol + 8], gK + 8);
      const _Float16* gV = Vglob + (size_t)vrow * 1024 + tBase + vcol;
      cp16_to_lds(&Vs[vrow * LDT + vcol],     gV);
      cp16_to_lds(&Vs[vrow * LDT + vcol + 8], gV + 8);
      async_wait0();
    }
    __syncthreads();
    if (tBase <= qm + 15) {              // wave-uniform causal skip
#pragma unroll
      for (int j = 0; j < 2; ++j) {
        const int tj = tBase + j * 16;
        _Float16 av[8];
        if (tj <= qm + 15) {
          v8f sc = {};
#pragma unroll
          for (int kc = 0; kc < 4; ++kc) {
            v16h bfr = load_bt_frag(&Ks[(j * 16) * 136 + kc * 32], 136);
            sc = wmma_f16(qf[kc], bfr, sc);
          }
#pragma unroll
          for (int r = 0; r < 8; ++r) {
            const int srow = qm + r + 8 * half;
            const int t    = tj + n15;
            av[r] = (_Float16)((t <= srow && sc[r] > 0.f) ? 1.f : 0.f);
          }
        } else {
#pragma unroll
          for (int r = 0; r < 8; ++r) av[r] = (_Float16)0.f;
        }
#pragma unroll
        for (int r = 0; r < 8; ++r)
          Attn[(wave * 16 + r + 8 * half) * LDT + j * 16 + n15] = av[r];
      }
      // wave-local LDS round-trip re-layouts attn (C-frag -> A-frag)
      v16h afr = load_a_frag(&Attn[wave * 16 * LDT], LDT);
#pragma unroll
      for (int f = 0; f < 8; ++f) {
        v16h bfr = load_bt_frag(&Vs[f * 16 * LDT], LDT);
        oacc[f] = wmma_f16(afr, bfr, oacc[f]);
      }
    }
  }

  // ternary quantize O, store flat [B,S,NH*HD] for the output GEMM
  _Float16* Obase = OQ + (size_t)b * 1024 * 2048;
#pragma unroll
  for (int f = 0; f < 8; ++f) {
    const int d = f * 16 + n15;
#pragma unroll
    for (int r = 0; r < 8; ++r) {
      const int s = qm + r + 8 * half;
      Obase[(size_t)s * 2048 + h * 128 + d] = (_Float16)tern(oacc[f][r]);
    }
  }
}

// ---------------------------------------------------------------------------
// Kernel 3: out = O @ Wo^T, ternary quantize, f32 output.
// grid=(16,16) block=256; WG tile 128x128, K=2048.
// ---------------------------------------------------------------------------
__global__ __launch_bounds__(256)
void gemm_out_kernel(const _Float16* __restrict__ O,
                     const _Float16* __restrict__ W,
                     float* __restrict__ OUT) {
  __shared__ alignas(16) _Float16 As[128 * LDT];
  __shared__ alignas(16) _Float16 Bs[128 * LDT];
  const int tid  = threadIdx.x;
  const int lane = tid & 31, wave = tid >> 5;
  const int mBase = blockIdx.x * 128;
  const int nBase = blockIdx.y * 128;
  const int lrow = tid >> 1;
  const int lcol = (tid & 1) * 16;

  v8f acc[8] = {};

  for (int k0 = 0; k0 < 2048; k0 += 32) {
    __syncthreads();
    {
      const _Float16* gA = O + (size_t)(mBase + lrow) * 2048 + k0 + lcol;
      cp16_to_lds(&As[lrow * LDT + lcol],     gA);
      cp16_to_lds(&As[lrow * LDT + lcol + 8], gA + 8);
      const _Float16* gB = W + (size_t)(nBase + lrow) * 2048 + k0 + lcol;
      cp16_to_lds(&Bs[lrow * LDT + lcol],     gB);
      cp16_to_lds(&Bs[lrow * LDT + lcol + 8], gB + 8);
      if (k0 + 32 < 2048) {
        __builtin_prefetch((const void*)(gA + 32), 0, 1);
        __builtin_prefetch((const void*)(gB + 32), 0, 1);
      }
      async_wait0();
    }
    __syncthreads();
    v16h a = load_a_frag(&As[wave * 16 * LDT], LDT);
#pragma unroll
    for (int f = 0; f < 8; ++f) {
      v16h b = load_bt_frag(&Bs[f * 16 * LDT], LDT);
      acc[f] = wmma_f16(a, b, acc[f]);
    }
  }

  const int n15 = lane & 15, half = lane >> 4;
#pragma unroll
  for (int f = 0; f < 8; ++f) {
    const int n = nBase + f * 16 + n15;
#pragma unroll
    for (int r = 0; r < 8; ++r) {
      const int tok = mBase + wave * 16 + r + 8 * half;
      OUT[(size_t)tok * 2048 + n] = tern(acc[f][r]);
    }
  }
}

// ---------------------------------------------------------------------------
extern "C" void kernel_launch(void* const* d_in, const int* in_sizes, int n_in,
                              void* d_out, int out_size, void* d_ws, size_t ws_size,
                              hipStream_t stream) {
  (void)in_sizes; (void)n_in; (void)out_size; (void)ws_size;
  const float* x  = (const float*)d_in[0];   // [2,1024,2048]
  const float* Wq = (const float*)d_in[1];   // [3072,2048]
  const float* Wo = (const float*)d_in[2];   // [2048,2048]
  // d_in[3] = mask: causal, reconstructed analytically in-kernel.

  char* ws = (char*)d_ws;                    // needs ~48 MB
  _Float16* X16 = (_Float16*)(ws + ((size_t)0  << 20));
  _Float16* W1  = (_Float16*)(ws + ((size_t)8  << 20));
  _Float16* WO  = (_Float16*)(ws + ((size_t)20 << 20));
  _Float16* QRp = (_Float16*)(ws + ((size_t)28 << 20));
  _Float16* KRp = (_Float16*)(ws + ((size_t)36 << 20));
  _Float16* VTp = (_Float16*)(ws + ((size_t)38 << 20));
  _Float16* OQp = (_Float16*)(ws + ((size_t)40 << 20));

  cvt_kernel<<<4194304 / 2048, 256, 0, stream>>>(x,  X16, 4194304);
  cvt_kernel<<<6291456 / 2048, 256, 0, stream>>>(Wq, W1,  6291456);
  cvt_kernel<<<4194304 / 2048, 256, 0, stream>>>(Wo, WO,  4194304);
  gemm_qkv_rope_kernel<<<dim3(16, 24), 256, 0, stream>>>(X16, W1, QRp, KRp, VTp);
  attention_kernel<<<dim3(8, 32), 256, 0, stream>>>(QRp, KRp, VTp, OQp);
  gemm_out_kernel<<<dim3(16, 16), 256, 0, stream>>>(OQp, WO, (float*)d_out);
}
